// SMN_52742198395065
// MI455X (gfx1250) — compile-verified, matrix operand order checked by reading, not run on hardware
//
#include <hip/hip_runtime.h>
#include <math.h>

#define NNODES 32768
#define NHOP   6
#define NHEADS 8
#define NHID   64
#define NFEAT  512
#define NPRE   512            // NHEADS*NHID
#define NSSF   256
#define NCLASS 40
#define NEG_SLOPE 0.2f

// ---- d_out segment offsets (f32 elements), tuple order:
// out [N,40], ssf_norm [256,40], z [N,256], spatial_loss [N,40], sigma [2]
#define OUT_OFF  0
#define SSF_OFF  (NNODES * NCLASS)                    // 1310720
#define Z_OFF    (SSF_OFF + NSSF * NCLASS)            // 1320960
#define SL_OFF   (Z_OFF + (size_t)NNODES * NSSF)      // 9709568
#define SIG_OFF  (SL_OFF + (size_t)NNODES * NCLASS)   // 11020288

// ---- workspace layout
#define WS_CSQ    0                                   // f32[40]
#define WS_CN     40                                  // f32[40]
#define WS_SSFNT  80                                  // f32[40][256]
#define WPRE_BYTE_OFF   65536                         // bf16 W_pre  [512][512] (512KB)
#define WPOST_BYTE_OFF  (WPRE_BYTE_OFF + NPRE * NFEAT * 2)   // 589824 (+256KB)
#define Z1_BYTE_OFF     1048576                       // bf16 z1 [NNODES][NPRE] (32MB)

typedef __attribute__((ext_vector_type(16))) __bf16        v16bf;
typedef __attribute__((ext_vector_type(2)))  __bf16        v2bf;
typedef __attribute__((ext_vector_type(8)))  float         v8f;
typedef __attribute__((ext_vector_type(8)))  unsigned int  v8u;

// Packed f32x2 -> bf16x2 using the hardware converter when available.
static __device__ __forceinline__ unsigned int pack_bf16x2(float lo, float hi) {
#if __has_builtin(__builtin_amdgcn_cvt_pk_bf16_f32)
  auto p = __builtin_amdgcn_cvt_pk_bf16_f32(lo, hi);
  return __builtin_bit_cast(unsigned int, p);
#else
  v2bf p;
  p[0] = (__bf16)lo;
  p[1] = (__bf16)hi;
  return __builtin_bit_cast(unsigned int, p);
#endif
}
static __device__ __forceinline__ unsigned short f32_to_bf16(float f) {
  return (unsigned short)pack_bf16x2(f, f);
}
static __device__ __forceinline__ float bf16_to_f32(unsigned short u) {
  return __uint_as_float(((unsigned int)u) << 16);
}
static __device__ __forceinline__ float lrelu(float v) {
  return (v >= 0.f) ? v : NEG_SLOPE * v;
}
// WMMA fragment (16 bf16 per lane): elements p[0..7] and p[16..23] along K.
static __device__ __forceinline__ v8u load_bf16_frag(const unsigned short* __restrict__ p) {
  uint4 lo = *(const uint4*)p;
  uint4 hi = *(const uint4*)(p + 16);
  v8u r = { lo.x, lo.y, lo.z, lo.w, hi.x, hi.y, hi.z, hi.w };
  return r;
}

// =====================================================================
// Kernel W: one-shot weight conversion f32 -> bf16 (hot-loop hoist)
// =====================================================================
__global__ __launch_bounds__(256) void kw_cvt(const float* __restrict__ Wpre,
                                              const float* __restrict__ Wpost,
                                              unsigned short* __restrict__ wpre_bf,
                                              unsigned short* __restrict__ wpost_bf) {
  const int npre = NPRE * NFEAT;    // 262144
  const int npost = NSSF * NPRE;    // 131072
  int i = (blockIdx.x * 256 + threadIdx.x) * 4;
  if (i < npre) {
    float4 f = *(const float4*)(Wpre + i);
    uint2 p; p.x = pack_bf16x2(f.x, f.y); p.y = pack_bf16x2(f.z, f.w);
    *(uint2*)(wpre_bf + i) = p;
  } else {
    int j = i - npre;
    if (j < npost) {
      float4 f = *(const float4*)(Wpost + j);
      uint2 p; p.x = pack_bf16x2(f.x, f.y); p.y = pack_bf16x2(f.z, f.w);
      *(uint2*)(wpost_bf + j) = p;
    }
  }
}

// =====================================================================
// Kernel A: ssf sparsify (radix-select threshold), normalize, col stats
// =====================================================================
__global__ __launch_bounds__(256) void ka_ssf(const float* __restrict__ ssf,
                                              const float* __restrict__ sigma,
                                              float* __restrict__ out,
                                              float* __restrict__ ws) {
  __shared__ float vals[NSSF * NCLASS];     // 10240 f32 = 40KB
  __shared__ unsigned int hist[256];
  __shared__ unsigned int sel_prefix, sel_rem;
  const int tid = threadIdx.x;

  for (int i = tid; i < NSSF * NCLASS; i += 256) vals[i] = fabsf(ssf[i]);
  if (tid == 0) { sel_prefix = 0u; sel_rem = 5120u; }  // round(10240*0.5)
  __syncthreads();

  // MSB-first radix select: value at ascending-sorted index 5120.
  for (int pass = 3; pass >= 0; --pass) {
    hist[tid] = 0;
    __syncthreads();
    const unsigned int shift = 8u * (unsigned)pass;
    const unsigned int pref = sel_prefix;
    for (int i = tid; i < NSSF * NCLASS; i += 256) {
      unsigned int key = __float_as_uint(vals[i]);
      bool match = (pass == 3) || ((key >> (shift + 8u)) == pref);
      if (match) atomicAdd(&hist[(key >> shift) & 255u], 1u);
    }
    __syncthreads();
    if (tid == 0) {
      unsigned int rem = sel_rem;
      unsigned int b = 0;
      for (; b < 256u; ++b) { if (rem < hist[b]) break; rem -= hist[b]; }
      sel_prefix = (pref << 8) | b;
      sel_rem = rem;
    }
    __syncthreads();
  }
  const float thr = __uint_as_float(sel_prefix);

  // Row-wise: sparsify + L1 normalize (reuse vals as ssf_norm storage)
  {
    float s = 0.f;
    for (int c = 0; c < NCLASS; ++c) {
      float v = ssf[tid * NCLASS + c];
      float kept = (fabsf(v) >= thr) ? v : 0.f;
      vals[tid * NCLASS + c] = kept;
      s += fabsf(kept);
    }
    float rs = 1.0f / fmaxf(s, 1e-12f);
    for (int c = 0; c < NCLASS; ++c) {
      float nv = vals[tid * NCLASS + c] * rs;
      vals[tid * NCLASS + c] = nv;
      out[SSF_OFF + tid * NCLASS + c] = nv;
    }
  }
  __syncthreads();

  // Column stats + transposed copy for contiguous dots in kernel C
  if (tid < NCLASS) {
    float csq = 0.f;
    for (int r = 0; r < NSSF; ++r) {
      float v = vals[r * NCLASS + tid];
      csq += v * v;
      ws[WS_SSFNT + tid * NSSF + r] = v;
    }
    ws[WS_CSQ + tid] = csq;
    ws[WS_CN + tid] = fmaxf(sqrtf(csq), 1e-6f);
  }
  if (tid < 2) out[SIG_OFF + tid] = sigma[tid];
}

// =====================================================================
// Kernel B: fused pre-GEMM (bf16 WMMA) + hop-attention softmax -> z1
// One WG (256 thr = 8 waves) handles 8 nodes = 48 GEMM rows x 512 cols.
// =====================================================================
#define BPAD    16
#define BSTRIDE (NFEAT + BPAD)   // 528 shorts per row (16B-aligned rows)

__global__ __launch_bounds__(256) void kb_pre(const float* __restrict__ x,
                                              const unsigned short* __restrict__ wpre_bf,
                                              const float* __restrict__ bpre,
                                              const float* __restrict__ att,
                                              unsigned short* __restrict__ z1) {
  __shared__ __align__(16) unsigned short hsm[48 * BSTRIDE];  // 50688 B
  __shared__ float attl[NHEADS * NHOP * NHID];                // 12288 B
  const int tid  = threadIdx.x;
  const int wave = tid >> 5;
  const int lane = tid & 31;
  const int sub  = lane & 15;
  const int hi   = lane >> 4;
  const int nodeBase = blockIdx.x * 8;

  for (int i = tid; i < NHEADS * NHOP * NHID; i += 256) attl[i] = att[i];

  // Stage 1: x tile [48,512] f32 -> bf16 in LDS (A-operand feed)
  const float* xin = x + (size_t)nodeBase * (NHOP * NFEAT);
  for (int e4 = tid * 4; e4 < 48 * NFEAT; e4 += 256 * 4) {
    float4 f = *(const float4*)(xin + e4);
    int row = e4 >> 9, col = e4 & 511;
    uint2 p; p.x = pack_bf16x2(f.x, f.y); p.y = pack_bf16x2(f.z, f.w);
    *(uint2*)&hsm[row * BSTRIDE + col] = p;
  }
  __syncthreads();

  // Stage 2: GEMM. Wave owns 64-col strip: 3 row-tiles x 4 col-tiles.
  const int colBase = wave * 64;
  v8f zero8 = {0.f, 0.f, 0.f, 0.f, 0.f, 0.f, 0.f, 0.f};
  v8f acc[3][4];
#pragma unroll
  for (int rt = 0; rt < 3; ++rt)
#pragma unroll
    for (int ct = 0; ct < 4; ++ct) acc[rt][ct] = zero8;

#pragma unroll 2
  for (int kb = 0; kb < NFEAT; kb += 32) {
    const int k0 = kb + hi * 8;
    // Issue ALL fragment loads up front so waits can be watermarked.
    v8u bfrag[4], afrag[3];
#pragma unroll
    for (int ct = 0; ct < 4; ++ct) {
      int o = colBase + ct * 16 + sub;
      bfrag[ct] = load_bf16_frag(wpre_bf + (size_t)o * NFEAT + k0);
    }
#pragma unroll
    for (int rt = 0; rt < 3; ++rt)
      afrag[rt] = load_bf16_frag(&hsm[(rt * 16 + sub) * BSTRIDE + k0]);
#pragma unroll
    for (int rt = 0; rt < 3; ++rt) {
      v16bf a = __builtin_bit_cast(v16bf, afrag[rt]);
#pragma unroll
      for (int ct = 0; ct < 4; ++ct) {
        acc[rt][ct] = __builtin_amdgcn_wmma_f32_16x16x32_bf16(
            false, a, false, __builtin_bit_cast(v16bf, bfrag[ct]),
            (short)0, acc[rt][ct], false, false);
      }
    }
  }
  __syncthreads();   // all waves done reading x-tile

  // Stage 3: bias + lrelu -> bf16 h back into the same LDS
#pragma unroll
  for (int ct = 0; ct < 4; ++ct) {
    int col = colBase + ct * 16 + sub;
    float b = bpre[col];
#pragma unroll
    for (int rt = 0; rt < 3; ++rt) {
#pragma unroll
      for (int j = 0; j < 8; ++j) {
        int row = rt * 16 + j + hi * 8;
        hsm[row * BSTRIDE + col] = f32_to_bf16(lrelu(acc[rt][ct][j] + b));
      }
    }
  }
  __syncthreads();

  // Stage 4: per (node, channel) softmax over 6 hops; z1 = sum h*w.
  // Each thread does two adjacent channels -> one packed b32 store.
  for (int idx = tid * 2; idx < 8 * NPRE; idx += 256 * 2) {
    const int node = idx >> 9;
    const int o    = idx & 511;          // even
    float zpair[2];
#pragma unroll
    for (int q = 0; q < 2; ++q) {
      const int oc = o + q;
      const float* av = &attl[(oc >> 6) * (NHOP * NHID) + (oc & 63)];
      const unsigned short* hrow = &hsm[(node * NHOP) * BSTRIDE + oc];
      float h[NHOP], sc[NHOP];
#pragma unroll
      for (int hop = 0; hop < NHOP; ++hop) h[hop] = bf16_to_f32(hrow[hop * BSTRIDE]);
      const float am0 = h[0] * av[0];
      float mx = -1e30f;
#pragma unroll
      for (int hop = 0; hop < NHOP; ++hop) {
        float s = lrelu(am0 + h[hop] * av[hop * NHID]);
        sc[hop] = s;
        mx = fmaxf(mx, s);
      }
      float sum = 0.f, zv = 0.f;
#pragma unroll
      for (int hop = 0; hop < NHOP; ++hop) {
        float e = __expf(sc[hop] - mx);
        sum += e;
        zv  += h[hop] * e;
      }
      zpair[q] = zv / sum;
    }
    *(unsigned int*)(z1 + (size_t)(nodeBase + node) * NPRE + o) =
        pack_bf16x2(zpair[0], zpair[1]);
  }
}

// =====================================================================
// Kernel C: post-GEMM (bf16 WMMA) + out/cdist/cos-sim + log-softmaxes
// One WG (256 thr = 8 waves) handles 32 rows x all 256 cols.
// =====================================================================
#define ZSS 520     // bf16 stage stride (shorts)
#define Z2S 264     // f32 z stride (floats)

__global__ __launch_bounds__(256) void kc_post(const unsigned short* __restrict__ z1,
                                               const unsigned short* __restrict__ wpost_bf,
                                               const float* __restrict__ bpost,
                                               const float* __restrict__ ws,
                                               float* __restrict__ out) {
  // zs (32x520 bf16 = 33280B) overlays z2s (32x264 f32 = 33792B)
  __shared__ __align__(16) unsigned char smem[32 * Z2S * 4];
  __shared__ float dists[32 * NCLASS], sims[32 * NCLASS];
  __shared__ float zsqs[32], zns[32], offd[32], offs[32];
  unsigned short* zs  = (unsigned short*)smem;
  float*          z2s = (float*)smem;

  const int tid  = threadIdx.x;
  const int wave = tid >> 5;
  const int lane = tid & 31;
  const int sub  = lane & 15;
  const int hi   = lane >> 4;
  const int rowBase = blockIdx.x * 32;

  // Stage 1: stage z1 tile [32,512] bf16 into LDS (pure copy, no cvt)
  const unsigned short* zin = z1 + (size_t)rowBase * NPRE;
  for (int e4 = tid * 4; e4 < 32 * NPRE; e4 += 256 * 4) {
    uint2 d = *(const uint2*)(zin + e4);
    int row = e4 >> 9, col = e4 & 511;
    *(uint2*)&zs[row * ZSS + col] = d;
  }
  __syncthreads();

  // Stage 2: GEMM: 2 row-tiles x 2 col-tiles per wave
  v8f zero8 = {0.f, 0.f, 0.f, 0.f, 0.f, 0.f, 0.f, 0.f};
  v8f acc[2][2];
#pragma unroll
  for (int rt = 0; rt < 2; ++rt)
#pragma unroll
    for (int i = 0; i < 2; ++i) acc[rt][i] = zero8;

#pragma unroll 2
  for (int kb = 0; kb < NPRE; kb += 32) {
    const int k0 = kb + hi * 8;
    v8u bfrag[2], afrag[2];
#pragma unroll
    for (int i = 0; i < 2; ++i) {
      int o = (wave * 2 + i) * 16 + sub;
      bfrag[i] = load_bf16_frag(wpost_bf + (size_t)o * NPRE + k0);
    }
#pragma unroll
    for (int rt = 0; rt < 2; ++rt)
      afrag[rt] = load_bf16_frag(&zs[(rt * 16 + sub) * ZSS + k0]);
#pragma unroll
    for (int rt = 0; rt < 2; ++rt) {
      v16bf a = __builtin_bit_cast(v16bf, afrag[rt]);
#pragma unroll
      for (int i = 0; i < 2; ++i) {
        acc[rt][i] = __builtin_amdgcn_wmma_f32_16x16x32_bf16(
            false, a, false, __builtin_bit_cast(v16bf, bfrag[i]),
            (short)0, acc[rt][i], false, false);
      }
    }
  }
  __syncthreads();   // done reading zs; may overwrite as z2s

  // Stage 3: bias + lrelu -> f32 z in LDS
#pragma unroll
  for (int i = 0; i < 2; ++i) {
    int col = (wave * 2 + i) * 16 + sub;
    float b = bpost[col];
#pragma unroll
    for (int rt = 0; rt < 2; ++rt) {
#pragma unroll
      for (int j = 0; j < 8; ++j) {
        int row = rt * 16 + j + hi * 8;
        z2s[row * Z2S + col] = lrelu(acc[rt][i][j] + b);
      }
    }
  }
  __syncthreads();

  // Stage 4: row norms + write z output
  if (tid < 32) {
    float s = 0.f;
#pragma unroll 4
    for (int c = 0; c < NSSF; ++c) { float v = z2s[tid * Z2S + c]; s += v * v; }
    zsqs[tid] = s;
    zns[tid]  = fmaxf(sqrtf(s), 1e-6f);
  }
  for (int e = tid; e < 32 * NSSF; e += 256) {
    int row = e >> 8, col = e & 255;
    out[Z_OFF + (size_t)(rowBase + row) * NSSF + col] = z2s[row * Z2S + col];
  }
  __syncthreads();

  // Stage 5: out = z @ ssf_norm, spatial dist & cosine sim
  const float* csq   = ws + WS_CSQ;
  const float* cn    = ws + WS_CN;
  const float* ssfnT = ws + WS_SSFNT;
  for (int p = tid; p < 32 * NCLASS; p += 256) {
    int row = p / NCLASS, cls = p % NCLASS;
    const float* fw = ssfnT + cls * NSSF;
    const float* zr = z2s + row * Z2S;
    float d = 0.f;
#pragma unroll 4
    for (int k = 0; k < NSSF; ++k) d += zr[k] * fw[k];
    out[OUT_OFF + (size_t)(rowBase + row) * NCLASS + cls] = d;
    float sq = zsqs[row] + csq[cls] - 2.f * d;
    dists[p] = -sqrtf(fmaxf(sq, 0.f));
    sims[p]  = d / (zns[row] * cn[cls]);
  }
  __syncthreads();

  // Stage 6: log-softmax offsets per row
  if (tid < 32) {
    float md = -1e30f, ms = -1e30f;
    for (int c = 0; c < NCLASS; ++c) {
      md = fmaxf(md, dists[tid * NCLASS + c]);
      ms = fmaxf(ms, sims[tid * NCLASS + c]);
    }
    float sd = 0.f, ss = 0.f;
    for (int c = 0; c < NCLASS; ++c) {
      sd += __expf(dists[tid * NCLASS + c] - md);
      ss += __expf(sims[tid * NCLASS + c] - ms);
    }
    offd[tid] = md + __logf(sd);
    offs[tid] = ms + __logf(ss);
  }
  __syncthreads();

  for (int p = tid; p < 32 * NCLASS; p += 256) {
    int row = p / NCLASS;
    out[SL_OFF + (size_t)rowBase * NCLASS + p] =
        0.5f * ((dists[p] - offd[row]) + (sims[p] - offs[row]));
  }
}

// =====================================================================
extern "C" void kernel_launch(void* const* d_in, const int* in_sizes, int n_in,
                              void* d_out, int out_size, void* d_ws, size_t ws_size,
                              hipStream_t stream) {
  const float* x     = (const float*)d_in[0];
  const float* Wpre  = (const float*)d_in[1];
  const float* bpre  = (const float*)d_in[2];
  const float* att   = (const float*)d_in[3];
  const float* Wpost = (const float*)d_in[4];
  const float* bpost = (const float*)d_in[5];
  const float* ssf   = (const float*)d_in[6];
  const float* sigma = (const float*)d_in[7];
  float* out = (float*)d_out;
  float* wsf = (float*)d_ws;
  unsigned short* wpre_bf  = (unsigned short*)((char*)d_ws + WPRE_BYTE_OFF);
  unsigned short* wpost_bf = (unsigned short*)((char*)d_ws + WPOST_BYTE_OFF);
  unsigned short* z1       = (unsigned short*)((char*)d_ws + Z1_BYTE_OFF);

  const int ncvt = (NPRE * NFEAT + NSSF * NPRE) / 4;   // 98304 threads
  kw_cvt <<<dim3(ncvt / 256),  dim3(256), 0, stream>>>(Wpre, Wpost, wpre_bf, wpost_bf);
  ka_ssf <<<dim3(1),           dim3(256), 0, stream>>>(ssf, sigma, out, wsf);
  kb_pre <<<dim3(NNODES / 8),  dim3(256), 0, stream>>>(x, wpre_bf, bpre, att, z1);
  kc_post<<<dim3(NNODES / 32), dim3(256), 0, stream>>>(z1, wpost_bf, bpost, wsf, out);
}
